// SelfAttentionV3_3848290697374
// MI455X (gfx1250) — matrix operand order repaired and checked by reading, not compiled, not run on hardware
//
#include <hip/hip_runtime.h>
#include <hip/hip_bf16.h>
#include <stdint.h>

// ---------------------------------------------------------------------------
// Causal self-attention for MI455X (gfx1250, wave32, WMMA + TDM).
//   N = 4096 tokens, D = 1024 model dim, fp32 in/out, bf16 WMMA compute.
// Stages:
//   1) cvt:   X -> bf16 (Xh),  Wq/Wk/Wv -> bf16 transposed (Wt[n][k])
//   2) gemm:  Q = X@Wq, K = X@Wk (row major bf16);  V stored transposed Vt[d][n]
//             A/B tiles staged in LDS by the Tensor Data Mover, double-buffered
//             with s_wait_tensorcnt; TDM pad feature gives 80B LDS row stride.
//   3) stats: per-row softmax max m and reciprocal sum 1/l  (streams K, WMMA S)
//   4) out:   O = (exp(S*sc - m)/l) @ V, P staged in LDS, WMMA everywhere
// Workspace layout (needs ~38.1 MB):
//   [0,8M)   Xh      bf16 [N][D]
//   [8M,14M) Wt      bf16 [3][D][D]   (transposed weights)
//   [14M,22M) Qh     bf16 [N][D]
//   [22M,30M) Kh     bf16 [N][D]
//   [30M,38M) Vt     bf16 [D][N]     (transposed V)
//   [38M, +32K) mbuf / rlbuf  f32 [N] each
// ---------------------------------------------------------------------------

#define ND 4096
#define DD 1024
#define SCALE 0.03125f  // 1/sqrt(1024)

typedef __attribute__((ext_vector_type(16))) __bf16        v16bf;
typedef __attribute__((ext_vector_type(8)))  float         v8f;
typedef __attribute__((ext_vector_type(4)))  int           v4i;
typedef __attribute__((ext_vector_type(8)))  int           v8i;
typedef __attribute__((ext_vector_type(4)))  unsigned int  v4u;

union Frag {
  v16bf v;
  v4i   q[2];
};

__device__ __forceinline__ unsigned short f32_to_bf16(float f) {
  unsigned int u = __float_as_uint(f);
  u += 0x7fffu + ((u >> 16) & 1u);  // round-to-nearest-even
  return (unsigned short)(u >> 16);
}

__device__ __forceinline__ v8f vzero8() {
  v8f z = {0.f, 0.f, 0.f, 0.f, 0.f, 0.f, 0.f, 0.f};
  return z;
}

// A-matrix 16x32 bf16 fragment (row-major source, 'rowptr' -> row (lane&15),
// already offset to k-base). Per-lane: elems 0..15 = K hi*16..hi*16+15
// (two 16B chunks).
__device__ __forceinline__ Frag load_a_frag(const unsigned short* rowptr, int hi) {
  Frag f;
  f.q[0] = *(const v4i*)(rowptr + hi * 16);
  f.q[1] = *(const v4i*)(rowptr + hi * 16 + 8);
  return f;
}

// B-matrix 32x16 bf16 fragment: per-lane 16 contiguous K elems at k-base+hi*16.
// 'p' must already point at (row n, kbase + hi*16).
__device__ __forceinline__ Frag load_b_frag(const unsigned short* p) {
  Frag f;
  f.q[0] = *(const v4i*)(p);
  f.q[1] = *(const v4i*)(p + 8);
  return f;
}

__device__ __forceinline__ v8f wmma_bf16(const Frag& a, const Frag& b, v8f c) {
  return __builtin_amdgcn_wmma_f32_16x16x32_bf16(false, a.v, false, b.v,
                                                 (short)0, c, false, false);
}

// ---------------------------------------------------------------- TDM ------
// Issue a TDM 2D tile load: 128 rows x 32 bf16 elems, row-major source with
// row stride DD elems, into LDS at lds_off with 16B padding per 64B row
// (-> 80B LDS row stride = 40 elems). Tracked by TENSORcnt.
#define TILE_ROWS   128
#define TILE_K      32
#define LDS_STRIDE  40   // elems per row in LDS after TDM padding (80 bytes)

__device__ __forceinline__ void tdm_load_tile(const unsigned short* gsrc,
                                              unsigned lds_off) {
  unsigned long long ga = (unsigned long long)(uintptr_t)gsrc;
  v4u g0;
  g0[0] = 1u;                                            // count=1, user mode
  g0[1] = lds_off;                                       // lds_addr
  g0[2] = (unsigned)(ga & 0xffffffffu);                  // global_addr[31:0]
  g0[3] = (unsigned)((ga >> 32) & 0x1ffffffu)            // global_addr[56:32]
          | (2u << 30);                                  // type = 2 (image)
  v8i g1;
  g1[0] = (1 << 16)     // data_size = 1 -> 2 bytes
        | (1 << 20)     // pad_enable
        | (3 << 22)     // pad_interval = 3 -> every 16 DWORDs (64B)
        | (3 << 25);    // pad_amount = 3 -> 4 DWORDs (16B)
  g1[1] = (DD & 0xffff) << 16;                           // tensor_dim0 lo16
  g1[2] = (DD >> 16) | ((ND & 0xffff) << 16);            // dim0 hi / dim1 lo
  g1[3] = (ND >> 16) | (TILE_K << 16);                   // dim1 hi / tile_dim0
  g1[4] = TILE_ROWS;                                     // tile_dim1 (tile_dim2=0)
  g1[5] = DD;                                            // tensor_dim0_stride lo32
  g1[6] = 0;                                             // stride0 hi / stride1 lo
  g1[7] = 0;
  v4i gz4 = {0, 0, 0, 0};                                // 2D: groups 2/3 unused
  v8i gz8 = {0, 0, 0, 0, 0, 0, 0, 0};
  __builtin_amdgcn_tensor_load_to_lds(g0, g1, gz4, gz4, gz8, 0);
}

// ---------------------------------------------------------------- converts --
__global__ void attn_cvt_x(const float* __restrict__ X,
                           unsigned short* __restrict__ Xh, int n) {
  int i = blockIdx.x * 256 + threadIdx.x;
  if (i < n) Xh[i] = f32_to_bf16(X[i]);
}

// W[k][n] (row-major [D][D]) -> Wt[n][k] bf16
__global__ void attn_cvt_wT(const float* __restrict__ W,
                            unsigned short* __restrict__ Wt) {
  int i = blockIdx.x * 256 + threadIdx.x;  // i < D*D
  int k = i >> 10;          // i / DD
  int n = i & (DD - 1);     // i % DD
  Wt[n * DD + k] = f32_to_bf16(W[i]);
}

// ------------------------------------------------------------------- gemm --
// grid: (D/128, N/128, 3). block 256 = 8 waves. Block tile 128(M) x 128(N).
// Wave tile 32x64 = 2M x 4N wmma tiles, K-loop step 32. A (X rows) and B
// (Wt rows) tiles staged in LDS by TDM, double-buffered.
__global__ void attn_qkv_gemm(const unsigned short* __restrict__ Xh,
                              const unsigned short* __restrict__ WtAll,
                              unsigned short* __restrict__ Qh,
                              unsigned short* __restrict__ Kh,
                              unsigned short* __restrict__ Vt) {
  __shared__ unsigned short ldsA[2][TILE_ROWS * LDS_STRIDE];
  __shared__ unsigned short ldsB[2][TILE_ROWS * LDS_STRIDE];

  const int w    = blockIdx.z;
  const unsigned short* Wt = WtAll + (size_t)w * DD * DD;
  const int tid  = threadIdx.x;
  const int wave = tid >> 5;
  const int lane = tid & 31;
  const int nl   = lane & 15;
  const int hi   = lane >> 4;
  const int mw   = (wave & 3) * 32;   // wave M offset inside block tile
  const int nw   = (wave >> 2) * 64;  // wave N offset inside block tile

  const unsigned short* Atile = Xh + (size_t)(blockIdx.y * 128) * DD;
  const unsigned short* Btile = Wt + (size_t)(blockIdx.x * 128) * DD;

  v8f acc[2][4];
#pragma unroll
  for (int i = 0; i < 2; ++i)
#pragma unroll
    for (int j = 0; j < 4; ++j) acc[i][j] = vzero8();

  // Prologue: wave 0 kicks off TDM loads for k=0 into buffer 0.
  if (wave == 0) {
    tdm_load_tile(Atile, (unsigned)(uintptr_t)&ldsA[0][0]);
    tdm_load_tile(Btile, (unsigned)(uintptr_t)&ldsB[0][0]);
  }

  int cur = 0;
  for (int k = 0; k < DD; k += 32) {
    const int nxt = cur ^ 1;
    if (wave == 0) {
      if (k + 32 < DD) {
        // Prefetch next tiles, then wait until the *current* pair landed
        // (TDM ops of one wave complete in order: <=2 outstanding means the
        // older pair is done).
        tdm_load_tile(Atile + (k + 32), (unsigned)(uintptr_t)&ldsA[nxt][0]);
        tdm_load_tile(Btile + (k + 32), (unsigned)(uintptr_t)&ldsB[nxt][0]);
        __builtin_amdgcn_s_wait_tensorcnt(2);
      } else {
        __builtin_amdgcn_s_wait_tensorcnt(0);
      }
    }
    __syncthreads();  // current tiles visible to all waves

    Frag a[2], b[4];
#pragma unroll
    for (int i = 0; i < 2; ++i)
      a[i] = load_a_frag(&ldsA[cur][(mw + i * 16 + nl) * LDS_STRIDE], hi);
#pragma unroll
    for (int j = 0; j < 4; ++j)
      b[j] = load_b_frag(&ldsB[cur][(nw + j * 16 + nl) * LDS_STRIDE + hi * 16]);
#pragma unroll
    for (int i = 0; i < 2; ++i)
#pragma unroll
      for (int j = 0; j < 4; ++j) acc[i][j] = wmma_bf16(a[i], b[j], acc[i][j]);

    __syncthreads();  // everyone done reading 'cur' before TDM reuses it
    cur = nxt;
  }

#pragma unroll
  for (int i = 0; i < 2; ++i)
#pragma unroll
    for (int j = 0; j < 4; ++j)
#pragma unroll
      for (int r = 0; r < 8; ++r) {
        int row = blockIdx.y * 128 + mw + i * 16 + hi * 8 + r;  // token index
        int col = blockIdx.x * 128 + nw + j * 16 + nl;          // feature index
        unsigned short h = f32_to_bf16(acc[i][j][r]);
        if (w == 0)      Qh[(size_t)row * DD + col] = h;
        else if (w == 1) Kh[(size_t)row * DD + col] = h;
        else             Vt[(size_t)col * ND + row] = h;  // transposed V
      }
}

// ------------------------------------------------------------------ stats --
// grid: N/128. block 256 = 8 waves; each wave owns 16 query rows and streams
// causal K blocks of 32 columns. Online (m, l) in registers; lane-half shuffles
// reduce across the 16 columns each half of the wave holds.
__global__ void attn_stats(const unsigned short* __restrict__ Qh,
                           const unsigned short* __restrict__ Kh,
                           float* __restrict__ mbuf,
                           float* __restrict__ rlbuf) {
  const int tid  = threadIdx.x;
  const int wave = tid >> 5;
  const int lane = tid & 31;
  const int nl   = lane & 15;
  const int hi   = lane >> 4;
  const int rowbase = blockIdx.x * 128 + wave * 16;
  const int maxrow  = rowbase + 15;

  float mrun[8], lrun[8];
#pragma unroll
  for (int r = 0; r < 8; ++r) { mrun[r] = -3.0e38f; lrun[r] = 0.f; }

  const unsigned short* qrow = Qh + (size_t)(rowbase + nl) * DD;

  for (int jb = 0; jb <= maxrow; jb += 32) {
    v8f s0 = vzero8(), s1 = vzero8();
    for (int k = 0; k < DD; k += 32) {
      Frag a  = load_a_frag(qrow + k, hi);
      Frag b0 = load_b_frag(Kh + (size_t)(jb + nl) * DD + k + hi * 16);
      Frag b1 = load_b_frag(Kh + (size_t)(jb + 16 + nl) * DD + k + hi * 16);
      s0 = wmma_bf16(a, b0, s0);
      s1 = wmma_bf16(a, b1, s1);
    }
#pragma unroll
    for (int r = 0; r < 8; ++r) {
      int row = rowbase + hi * 8 + r;
      float v0 = (jb + nl      <= row) ? s0[r] * SCALE : -3.0e38f;
      float v1 = (jb + 16 + nl <= row) ? s1[r] * SCALE : -3.0e38f;
      float mx = fmaxf(v0, v1);
#pragma unroll
      for (int off = 1; off < 16; off <<= 1)
        mx = fmaxf(mx, __shfl_xor(mx, off, 32));  // stays within 16-lane half
      float mnew = fmaxf(mrun[r], mx);
      float ps = __expf(v0 - mnew) + __expf(v1 - mnew);
#pragma unroll
      for (int off = 1; off < 16; off <<= 1)
        ps += __shfl_xor(ps, off, 32);
      lrun[r] = lrun[r] * __expf(mrun[r] - mnew) + ps;
      mrun[r] = mnew;
    }
  }

  if (nl == 0) {
#pragma unroll
    for (int r = 0; r < 8; ++r) {
      int row = rowbase + hi * 8 + r;
      mbuf[row]  = mrun[r];
      rlbuf[row] = 1.0f / lrun[r];  // diagonal guarantees lrun > 0
    }
  }
}

// -------------------------------------------------------------------- out --
// grid: (4 d-chunks of 256, 64 q-tiles of 64). block 256 = 8 waves.
// Per KV block of 32: each wave computes one 16x16 S tile (4M x 2N layout),
// converts to normalized P (bf16) in LDS, then all waves do P@V with
// A-fragments from LDS and B-fragments from transposed V in global.
#define P_STRIDE 40  // elems; 80B rows spread LDS banks
__global__ void attn_out(const unsigned short* __restrict__ Qh,
                         const unsigned short* __restrict__ Kh,
                         const unsigned short* __restrict__ Vt,
                         const float* __restrict__ mbuf,
                         const float* __restrict__ rlbuf,
                         float* __restrict__ Out) {
  __shared__ unsigned short Pl[64 * P_STRIDE];

  const int tid  = threadIdx.x;
  const int wave = tid >> 5;
  const int lane = tid & 31;
  const int nl   = lane & 15;
  const int hi   = lane >> 4;

  const int qbase = blockIdx.y * 64;
  const int d0    = blockIdx.x * 256;

  const int sm = wave & 3;   // S tile: 4 along M
  const int sn = wave >> 2;  // 2 along N (32 cols)
  const int om = wave & 1;   // O tile block: 2x32 rows
  const int on = wave >> 1;  // 4x64 cols

  v8f oacc[2][4];
#pragma unroll
  for (int i = 0; i < 2; ++i)
#pragma unroll
    for (int j = 0; j < 4; ++j) oacc[i][j] = vzero8();

  float mrow[8], rl[8];
#pragma unroll
  for (int r = 0; r < 8; ++r) {
    int row = qbase + sm * 16 + hi * 8 + r;
    mrow[r] = mbuf[row];
    rl[r]   = rlbuf[row];
  }

  const unsigned short* qrow = Qh + (size_t)(qbase + sm * 16 + nl) * DD;
  const int jmax = qbase + 63;

  for (int jb = 0; jb <= jmax; jb += 32) {
    // ---- S tile (16x16) for this wave, full-D reduction --------------------
    v8f s = vzero8();
    for (int k = 0; k < DD; k += 32) {
      Frag a = load_a_frag(qrow + k, hi);
      Frag b = load_b_frag(Kh + (size_t)(jb + sn * 16 + nl) * DD + k + hi * 16);
      s = wmma_bf16(a, b, s);
    }
    // ---- P = exp(S*sc - m)/l, causal-masked, staged in LDS ----------------
#pragma unroll
    for (int r = 0; r < 8; ++r) {
      int row = qbase + sm * 16 + hi * 8 + r;
      int col = jb + sn * 16 + nl;
      float p = (col <= row) ? __expf(s[r] * SCALE - mrow[r]) * rl[r] : 0.f;
      Pl[(sm * 16 + hi * 8 + r) * P_STRIDE + sn * 16 + nl] = f32_to_bf16(p);
    }
    __syncthreads();
    // ---- O += P @ V  (k = 32 = this j block) ------------------------------
    Frag pa[2];
#pragma unroll
    for (int i = 0; i < 2; ++i) {
      const unsigned short* pr = &Pl[(om * 32 + i * 16 + nl) * P_STRIDE];
      pa[i].q[0] = *(const v4i*)(pr + hi * 16);
      pa[i].q[1] = *(const v4i*)(pr + hi * 16 + 8);
    }
    Frag vb[4];
#pragma unroll
    for (int j = 0; j < 4; ++j) {
      int dcol = d0 + on * 64 + j * 16 + nl;  // Vt row = feature index
      vb[j] = load_b_frag(Vt + (size_t)dcol * ND + jb + hi * 16);
    }
#pragma unroll
    for (int i = 0; i < 2; ++i)
#pragma unroll
      for (int j = 0; j < 4; ++j) oacc[i][j] = wmma_bf16(pa[i], vb[j], oacc[i][j]);
    __syncthreads();
  }

#pragma unroll
  for (int i = 0; i < 2; ++i)
#pragma unroll
    for (int j = 0; j < 4; ++j)
#pragma unroll
      for (int r = 0; r < 8; ++r) {
        int row = qbase + om * 32 + i * 16 + hi * 8 + r;
        int col = d0 + on * 64 + j * 16 + nl;
        Out[(size_t)row * DD + col] = oacc[i][j][r];
      }
}

// ---------------------------------------------------------------------------
extern "C" void kernel_launch(void* const* d_in, const int* in_sizes, int n_in,
                              void* d_out, int out_size, void* d_ws, size_t ws_size,
                              hipStream_t stream) {
  (void)in_sizes; (void)n_in; (void)out_size; (void)ws_size;
  const float* X  = (const float*)d_in[0];
  const float* Wq = (const float*)d_in[1];
  const float* Wk = (const float*)d_in[2];
  const float* Wv = (const float*)d_in[3];
  float* Out = (float*)d_out;

  char* ws = (char*)d_ws;
  unsigned short* Xh = (unsigned short*)(ws);
  unsigned short* Wt = (unsigned short*)(ws + ((size_t)8  << 20));
  unsigned short* Qh = (unsigned short*)(ws + ((size_t)14 << 20));
  unsigned short* Kh = (unsigned short*)(ws + ((size_t)22 << 20));
  unsigned short* Vt = (unsigned short*)(ws + ((size_t)30 << 20));
  float* mbuf  = (float*)(ws + ((size_t)38 << 20));
  float* rlbuf = mbuf + ND;

  attn_cvt_x<<<(ND * DD) / 256, 256, 0, stream>>>(X, Xh, ND * DD);
  attn_cvt_wT<<<(DD * DD) / 256, 256, 0, stream>>>(Wq, Wt);
  attn_cvt_wT<<<(DD * DD) / 256, 256, 0, stream>>>(Wk, Wt + (size_t)DD * DD);
  attn_cvt_wT<<<(DD * DD) / 256, 256, 0, stream>>>(Wv, Wt + (size_t)2 * DD * DD);

  attn_qkv_gemm<<<dim3(DD / 128, ND / 128, 3), 256, 0, stream>>>(Xh, Wt, Qh, Kh, Vt);

  attn_stats<<<ND / 128, 256, 0, stream>>>(Qh, Kh, mbuf, rlbuf);

  attn_out<<<dim3(4, ND / 64), 256, 0, stream>>>(Qh, Kh, Vt, mbuf, rlbuf, Out);
}
